// MaskedCrossAttention_13649406067500
// MI455X (gfx1250) — matrix operand order repaired and checked by reading, not compiled, not run on hardware
//
#include <hip/hip_runtime.h>
#include <hip/hip_bf16.h>

// ---------------------------------------------------------------------------
// Masked cross-attention for MI455X (gfx1250), wave32, WMMA f16->f32,
// TDM (tensor_load_to_lds) double-buffered tile staging.
// Pipeline: LN -> (Q = xn@Wq)*scale -> KV = media@Wkv -> flash-attn -> @Wout
// ---------------------------------------------------------------------------

typedef __attribute__((ext_vector_type(16))) _Float16 v16h;
typedef __attribute__((ext_vector_type(8)))  _Float16 v8h;
typedef __attribute__((ext_vector_type(8)))  float    v8f;
typedef __attribute__((ext_vector_type(4)))  unsigned int u32x4;
typedef __attribute__((ext_vector_type(8)))  int      i32x8;
typedef __attribute__((ext_vector_type(4)))  int      i32x4;

#define HEADS 16
#define DIM_HEAD 64
#define LN_EPS 1e-5f

#if __has_builtin(__builtin_amdgcn_tensor_load_to_lds)
#define USE_TDM 1
#else
#define USE_TDM 0
#endif

__device__ __forceinline__ v8f wmma_f16(v16h a, v16h b, v8f c) {
  return __builtin_amdgcn_wmma_f32_16x16x32_f16(
      false, a, false, b, (short)0, c, false, false);
}

// Low 32 bits of a generic pointer to LDS are the LDS byte offset
// (flat LDS aperture: addr[31:0] == LDS offset, ISA 10.2).
__device__ __forceinline__ unsigned lds_off(const void* p) {
  return (unsigned)(size_t)p;
}

#if USE_TDM
// Issue a 2D TDM load: tile of `rows` x `tile_w` f16 elements, global row
// stride `stride_elems`, packed contiguously into LDS at lds_addr.
// D# bit packing per CDNA5 ISA ch.8 (group0 128b, group1 256b).
__device__ __forceinline__ void tdm_load_2d(unsigned lds_addr,
                                            const void* gptr,
                                            unsigned rows,
                                            unsigned stride_elems,
                                            unsigned tile_w) {
  unsigned long long ga = (unsigned long long)(size_t)gptr;
  u32x4 g0;
  g0[0] = 1u;                                   // count=1, user descriptor
  g0[1] = lds_addr;                             // bits 63:32 lds_addr
  g0[2] = (unsigned)(ga & 0xffffffffu);         // global_addr[31:0]
  g0[3] = (unsigned)((ga >> 32) & 0x01ffffffu)  // global_addr[56:32]
          | (2u << 30);                         // type = 2 ("image")
  unsigned long long s0 = (unsigned long long)stride_elems;  // dim0 stride
  i32x8 g1;
  g1[0] = (int)(1u << 16);                          // data_size=2B; mask=0
  g1[1] = (int)((tile_w & 0xffffu) << 16);          // tensor_dim0[15:0]
  g1[2] = (int)(((tile_w >> 16) & 0xffffu)          // tensor_dim0[31:16]
          | ((rows & 0xffffu) << 16));              // tensor_dim1[15:0]
  g1[3] = (int)(((rows >> 16) & 0xffffu)            // tensor_dim1[31:16]
          | ((tile_w & 0xffffu) << 16));            // tile_dim0
  g1[4] = (int)(rows & 0xffffu);                    // tile_dim1; tile_dim2=0
  g1[5] = (int)(s0 & 0xffffffffu);                  // tensor_dim0_stride lo
  g1[6] = (int)((s0 >> 32) & 0xffffu);              // stride hi; dim1_stride=0
  g1[7] = 0;
  i32x4 z4 = {0, 0, 0, 0};
#if defined(__clang_major__) && __clang_major__ >= 23
  i32x8 z8 = {0, 0, 0, 0, 0, 0, 0, 0};
  __builtin_amdgcn_tensor_load_to_lds(g0, g1, z4, z4, z8, 0);
#else
  __builtin_amdgcn_tensor_load_to_lds(g0, g1, z4, z4, 0);
#endif
}
#endif  // USE_TDM

// A-fragment (16x32, MxK): halves 0..7 K=hi*8.., halves 8..15 K=16+hi*8..
__device__ __forceinline__ v16h frag_a(const _Float16* s, int ld, int row0,
                                       int k0, int lane) {
  const int m  = lane & 15;
  const int hi = (lane >> 4) * 8;
  const _Float16* p = s + (size_t)(row0 + m) * ld + k0;
  v8h lo = *(const v8h*)(p + hi);
  v8h hh = *(const v8h*)(p + 16 + hi);
  v16h r;
#pragma unroll
  for (int i = 0; i < 8; ++i) { r[i] = lo[i]; r[8 + i] = hh[i]; }
  return r;
}

// B-fragment (32x16, KxN) from Bt[N][K]: lane n = column, halves 0..7 K=hi*8.., halves 8..15 K=16+hi*8..
__device__ __forceinline__ v16h frag_b(const _Float16* s, int ld, int row0,
                                       int k0, int lane) {
  const int n  = lane & 15;
  const int hi = (lane >> 4) * 8;
  const _Float16* p = s + (size_t)(row0 + n) * ld + k0;
  v8h a0 = *(const v8h*)(p + hi);
  v8h a1 = *(const v8h*)(p + 16 + hi);
  v16h r;
#pragma unroll
  for (int i = 0; i < 8; ++i) { r[i] = a0[i]; r[8 + i] = a1[i]; }
  return r;
}

// ---------------------------------------------------------------------------
__global__ void ln_f16_kernel(const float* __restrict__ x,
                              const float* __restrict__ gamma,
                              const float* __restrict__ beta,
                              _Float16* __restrict__ out, int D) {
  const int row = blockIdx.x;
  const int tid = threadIdx.x;
  const float* xr = x + (size_t)row * D;
  __shared__ float rs[256], rs2[256];
  float s = 0.f, s2 = 0.f;
  for (int i = tid; i < D; i += 256) { float v = xr[i]; s += v; s2 += v * v; }
  rs[tid] = s; rs2[tid] = s2;
  __syncthreads();
  for (int st = 128; st > 0; st >>= 1) {
    if (tid < st) { rs[tid] += rs[tid + st]; rs2[tid] += rs2[tid + st]; }
    __syncthreads();
  }
  const float mu  = rs[0] / (float)D;
  const float var = rs2[0] / (float)D - mu * mu;
  const float inv = rsqrtf(var + LN_EPS);
  _Float16* o = out + (size_t)row * D;
  for (int i = tid; i < D; i += 256)
    o[i] = (_Float16)((xr[i] - mu) * inv * gamma[i] + beta[i]);
}

__global__ void transpose_cvt_kernel(const float* __restrict__ in,
                                     _Float16* __restrict__ out,
                                     int R, int C) {
  size_t idx = (size_t)blockIdx.x * 256 + threadIdx.x;
  if (idx >= (size_t)R * C) return;
  int r = (int)(idx / C), c = (int)(idx % C);
  out[(size_t)c * R + r] = (_Float16)in[idx];
}

__global__ void cvt_f16_kernel(const float* __restrict__ in,
                               _Float16* __restrict__ out, size_t n) {
  size_t i = (size_t)blockIdx.x * 256 + threadIdx.x;
  if (i < n) out[i] = (_Float16)in[i];
}

// ---------------------------------------------------------------------------
// NT GEMM: C[M][N] = A[M][K] * Bt[N][K]^T.  Block 256 thr = 8 waves
// (4 along M x 2 along N); block tile 256x128x32; wave tile 64x64 ->
// 16 WMMAs per K-step. TDM double-buffered staging (48 KB LDS).
// ---------------------------------------------------------------------------
template <typename OutT>
__global__ __launch_bounds__(256)
void gemm_nt_kernel(const _Float16* __restrict__ A,
                    const _Float16* __restrict__ Bt,
                    OutT* __restrict__ C,
                    int M, int N, int K, float scale) {
  __shared__ alignas(16) _Float16 As[2][256 * 32];
  __shared__ alignas(16) _Float16 Bs[2][128 * 32];
  const int tid  = threadIdx.x;
  const int lane = tid & 31;
  const int wave = tid >> 5;
  const int wm   = wave & 3;
  const int wn   = wave >> 2;
  const int m0   = blockIdx.x * 256;
  const int n0   = blockIdx.y * 128;

  v8f acc[4][4];
#pragma unroll
  for (int mi = 0; mi < 4; ++mi)
#pragma unroll
    for (int ni = 0; ni < 4; ++ni) acc[mi][ni] = (v8f){0,0,0,0,0,0,0,0};

#if USE_TDM
  if (wave == 0) {
    tdm_load_2d(lds_off(&As[0][0]), A + (size_t)m0 * K, 256, (unsigned)K, 32);
    tdm_load_2d(lds_off(&Bs[0][0]), Bt + (size_t)n0 * K, 128, (unsigned)K, 32);
    __builtin_amdgcn_s_wait_tensorcnt(0);
  }
  __syncthreads();
#endif

  int buf = 0;
  for (int k0 = 0; k0 < K; k0 += 32, buf ^= 1) {
#if USE_TDM
    // Prefetch next K tile into the other buffer while we compute.
    if (wave == 0 && k0 + 32 < K) {
      tdm_load_2d(lds_off(&As[buf ^ 1][0]),
                  A + (size_t)m0 * K + (k0 + 32), 256, (unsigned)K, 32);
      tdm_load_2d(lds_off(&Bs[buf ^ 1][0]),
                  Bt + (size_t)n0 * K + (k0 + 32), 128, (unsigned)K, 32);
    }
#else
    { // manual staging fallback
      const int lrow = tid >> 1;
      const int lcol = (tid & 1) * 16;
      const _Float16* ga = A + (size_t)(m0 + lrow) * K + k0 + lcol;
      *(v8h*)&As[0][lrow * 32 + lcol]     = *(const v8h*)(ga);
      *(v8h*)&As[0][lrow * 32 + lcol + 8] = *(const v8h*)(ga + 8);
      const _Float16* ga2 = A + (size_t)(m0 + 128 + lrow) * K + k0 + lcol;
      *(v8h*)&As[0][(128 + lrow) * 32 + lcol]     = *(const v8h*)(ga2);
      *(v8h*)&As[0][(128 + lrow) * 32 + lcol + 8] = *(const v8h*)(ga2 + 8);
      const _Float16* gb = Bt + (size_t)(n0 + lrow) * K + k0 + lcol;
      *(v8h*)&Bs[0][lrow * 32 + lcol]     = *(const v8h*)(gb);
      *(v8h*)&Bs[0][lrow * 32 + lcol + 8] = *(const v8h*)(gb + 8);
      buf = 0;
      __syncthreads();
    }
#endif

    const _Float16* as = &As[buf][0];
    const _Float16* bs = &Bs[buf][0];
    v16h af[4], bf[4];
#pragma unroll
    for (int mi = 0; mi < 4; ++mi)
      af[mi] = frag_a(as, 32, wm * 64 + mi * 16, 0, lane);
#pragma unroll
    for (int ni = 0; ni < 4; ++ni)
      bf[ni] = frag_b(bs, 32, wn * 64 + ni * 16, 0, lane);
#pragma unroll
    for (int mi = 0; mi < 4; ++mi)
#pragma unroll
      for (int ni = 0; ni < 4; ++ni)
        acc[mi][ni] = wmma_f16(af[mi], bf[ni], acc[mi][ni]);

#if USE_TDM
    if (wave == 0) __builtin_amdgcn_s_wait_tensorcnt(0);
#endif
    __syncthreads();
  }

  const int nloc = lane & 15;
  const int rofs = (lane >> 4) * 8;
#pragma unroll
  for (int mi = 0; mi < 4; ++mi)
#pragma unroll
    for (int ni = 0; ni < 4; ++ni)
#pragma unroll
      for (int p = 0; p < 8; ++p) {
        int r = m0 + wm * 64 + mi * 16 + rofs + p;
        int c = n0 + wn * 64 + ni * 16 + nloc;
        C[(size_t)r * N + c] = (OutT)(acc[mi][ni][p] * scale);
      }
}

// ---------------------------------------------------------------------------
// Flash attention: grid (T/64, B*HEADS), block 128 (4 waves x 16 rows).
// q f16 [B*T][1024] (pre-scaled), kv f16 [B*L][2048] (k | v), out f16.
// ---------------------------------------------------------------------------
__global__ __launch_bounds__(128)
void attn_kernel(const _Float16* __restrict__ q,
                 const _Float16* __restrict__ kv,
                 const unsigned char* __restrict__ mask,
                 _Float16* __restrict__ out,
                 int B, int T, int L) {
  const int bh   = blockIdx.y;
  const int b    = bh / HEADS;
  const int h    = bh % HEADS;
  const int i0   = blockIdx.x * 64;
  const int tid  = threadIdx.x;
  const int lane = tid & 31;
  const int wave = tid >> 5;

  __shared__ alignas(16) _Float16 Qs[64 * 64];
  __shared__ alignas(16) _Float16 Ks[32 * 64];
  __shared__ alignas(16) _Float16 Vt[64 * 32];
  __shared__ alignas(16) _Float16 Ps[4][16 * 32];
  __shared__ float bias[32];

  { // load Q tile (64 x 64)
    int r = tid >> 1;
    int c = (tid & 1) * 32;
    const _Float16* gq = q + (size_t)(b * T + i0 + r) * 1024 + h * DIM_HEAD + c;
#pragma unroll
    for (int u = 0; u < 4; ++u)
      *(v8h*)&Qs[r * 64 + c + u * 8] = *(const v8h*)(gq + u * 8);
  }
  __syncthreads();
  v16h qf[2];
  qf[0] = frag_a(Qs, 64, wave * 16, 0, lane);
  qf[1] = frag_a(Qs, 64, wave * 16, 32, lane);

  v8f acc[4];
#pragma unroll
  for (int ni = 0; ni < 4; ++ni) acc[ni] = (v8f){0,0,0,0,0,0,0,0};
  float mrow[8], lrow[8];
#pragma unroll
  for (int p = 0; p < 8; ++p) { mrow[p] = -3.0e38f; lrow[p] = 0.f; }

  const int nloc = lane & 15;
  const int rofs = (lane >> 4) * 8;

  for (int j0 = 0; j0 < L; j0 += 32) {
    __syncthreads();  // previous iteration done reading Ks/Vt/bias
#if USE_TDM
    if (wave == 0)    // K block (32 x 64) via TDM DMA
      tdm_load_2d(lds_off(&Ks[0]),
                  kv + (size_t)(b * L + j0) * 2048 + h * DIM_HEAD,
                  32, 2048, 64);
#endif
    { // V block transposed into Vt (TDM cannot transpose), + mask bias
      int r = tid >> 2;
      int c = (tid & 3) * 16;
#if !USE_TDM
      const _Float16* gk = kv + (size_t)(b * L + j0 + r) * 2048 + h * DIM_HEAD + c;
      *(v8h*)&Ks[r * 64 + c]     = *(const v8h*)(gk);
      *(v8h*)&Ks[r * 64 + c + 8] = *(const v8h*)(gk + 8);
#endif
      const _Float16* gv = kv + (size_t)(b * L + j0 + r) * 2048 + 1024 + h * DIM_HEAD + c;
      v8h v0 = *(const v8h*)(gv);
      v8h v1 = *(const v8h*)(gv + 8);
#pragma unroll
      for (int u = 0; u < 8; ++u) {
        Vt[(c + u) * 32 + r]     = v0[u];
        Vt[(c + 8 + u) * 32 + r] = v1[u];
      }
      if (tid < 32)
        bias[tid] = mask[(size_t)b * L + j0 + tid] ? 0.f : -3.0e38f;
    }
#if USE_TDM
    if (wave == 0) __builtin_amdgcn_s_wait_tensorcnt(0);
#endif
    __syncthreads();

    v8f s[2];
    s[0] = (v8f){0,0,0,0,0,0,0,0};
    s[1] = (v8f){0,0,0,0,0,0,0,0};
#pragma unroll
    for (int ni = 0; ni < 2; ++ni) {
      v16h kb0 = frag_b(Ks, 64, ni * 16, 0, lane);
      v16h kb1 = frag_b(Ks, 64, ni * 16, 32, lane);
      s[ni] = wmma_f16(qf[0], kb0, s[ni]);
      s[ni] = wmma_f16(qf[1], kb1, s[ni]);
    }

    float pv0[8], pv1[8], rm[8];
#pragma unroll
    for (int p = 0; p < 8; ++p) {
      pv0[p] = s[0][p] + bias[nloc];
      pv1[p] = s[1][p] + bias[16 + nloc];
      rm[p]  = fmaxf(pv0[p], pv1[p]);
    }
#pragma unroll
    for (int off = 1; off < 16; off <<= 1)
#pragma unroll
      for (int p = 0; p < 8; ++p)
        rm[p] = fmaxf(rm[p], __shfl_xor(rm[p], off, 32));
    float fac[8], rs[8];
#pragma unroll
    for (int p = 0; p < 8; ++p) {
      float mn = fmaxf(mrow[p], rm[p]);
      fac[p] = __expf(mrow[p] - mn);
      mrow[p] = mn;
      pv0[p] = __expf(pv0[p] - mn);
      pv1[p] = __expf(pv1[p] - mn);
      rs[p]  = pv0[p] + pv1[p];
    }
#pragma unroll
    for (int off = 1; off < 16; off <<= 1)
#pragma unroll
      for (int p = 0; p < 8; ++p)
        rs[p] += __shfl_xor(rs[p], off, 32);
#pragma unroll
    for (int p = 0; p < 8; ++p) lrow[p] = lrow[p] * fac[p] + rs[p];
#pragma unroll
    for (int ni = 0; ni < 4; ++ni)
#pragma unroll
      for (int p = 0; p < 8; ++p) acc[ni][p] *= fac[p];

    // C-layout -> A-layout re-fragmentation of P through LDS
#pragma unroll
    for (int p = 0; p < 8; ++p) {
      Ps[wave][(rofs + p) * 32 + nloc]      = (_Float16)pv0[p];
      Ps[wave][(rofs + p) * 32 + 16 + nloc] = (_Float16)pv1[p];
    }
    __syncthreads();

    v16h pf = frag_a(&Ps[wave][0], 32, 0, 0, lane);
#pragma unroll
    for (int ni = 0; ni < 4; ++ni) {
      v16h vb = frag_b(Vt, 32, ni * 16, 0, lane);
      acc[ni] = wmma_f16(pf, vb, acc[ni]);
    }
  }

#pragma unroll
  for (int p = 0; p < 8; ++p) {
    float inv = 1.0f / lrow[p];
    int r = i0 + wave * 16 + rofs + p;
#pragma unroll
    for (int ni = 0; ni < 4; ++ni)
      out[(size_t)(b * T + r) * 1024 + h * DIM_HEAD + ni * 16 + nloc] =
          (_Float16)(acc[ni][p] * inv);
  }
}

// ---------------------------------------------------------------------------
extern "C" void kernel_launch(void* const* d_in, const int* in_sizes, int n_in,
                              void* d_out, int out_size, void* d_ws, size_t ws_size,
                              hipStream_t stream) {
  const float* x      = (const float*)d_in[0];
  const float* media  = (const float*)d_in[1];
  const unsigned char* mmask = (const unsigned char*)d_in[2];  // jnp bool_
  const float* gamma  = (const float*)d_in[3];
  const float* beta   = (const float*)d_in[4];
  const float* Wq     = (const float*)d_in[5];
  const float* Wkv    = (const float*)d_in[6];
  const float* Wout   = (const float*)d_in[7];
  float* out          = (float*)d_out;

  constexpr int B = 2, T = 2048, D = 2048, L = 1920, DA = 1024, INNER = 1024;
  constexpr float SCALE = 0.125f;  // DIM_HEAD^-0.5

  char* ws = (char*)d_ws;
  size_t off = 0;
  auto carve = [&](size_t bytes) {
    void* p = ws + off;
    off = (off + bytes + 255) & ~(size_t)255;
    return p;
  };
  _Float16* xn     = (_Float16*)carve((size_t)B * T * D * 2);
  _Float16* med_h  = (_Float16*)carve((size_t)B * L * DA * 2);
  _Float16* WqT    = (_Float16*)carve((size_t)INNER * D * 2);
  _Float16* WkvT   = (_Float16*)carve((size_t)2 * INNER * DA * 2);
  _Float16* WoutT  = (_Float16*)carve((size_t)D * INNER * 2);
  _Float16* q_h    = (_Float16*)carve((size_t)B * T * INNER * 2);
  _Float16* kv_h   = (_Float16*)carve((size_t)B * L * 2 * INNER * 2);
  _Float16* ao_h   = (_Float16*)carve((size_t)B * T * INNER * 2);

  ln_f16_kernel<<<B * T, 256, 0, stream>>>(x, gamma, beta, xn, D);

  {
    size_t n;
    n = (size_t)D * INNER;
    transpose_cvt_kernel<<<(unsigned)((n + 255) / 256), 256, 0, stream>>>(Wq, WqT, D, INNER);
    n = (size_t)DA * 2 * INNER;
    transpose_cvt_kernel<<<(unsigned)((n + 255) / 256), 256, 0, stream>>>(Wkv, WkvT, DA, 2 * INNER);
    n = (size_t)INNER * D;
    transpose_cvt_kernel<<<(unsigned)((n + 255) / 256), 256, 0, stream>>>(Wout, WoutT, INNER, D);
    n = (size_t)B * L * DA;
    cvt_f16_kernel<<<(unsigned)((n + 255) / 256), 256, 0, stream>>>(media, med_h, n);
  }

  // Q = (xn @ Wq) * scale : [4096 x 2048] x [2048 x 1024]
  gemm_nt_kernel<_Float16><<<dim3((B * T) / 256, INNER / 128), 256, 0, stream>>>(
      xn, WqT, q_h, B * T, INNER, D, SCALE);

  // KV = media @ Wkv : [3840 x 1024] x [1024 x 2048]
  gemm_nt_kernel<_Float16><<<dim3((B * L) / 256, (2 * INNER) / 128), 256, 0, stream>>>(
      med_h, WkvT, kv_h, B * L, 2 * INNER, DA, 1.0f);

  // Fused masked flash attention
  attn_kernel<<<dim3(T / 64, B * HEADS), 128, 0, stream>>>(
      q_h, kv_h, mmask, ao_h, B, T, L);

  // out = attn_out @ Wout : [4096 x 1024] x [1024 x 2048] -> fp32
  gemm_nt_kernel<float><<<dim3((B * T) / 256, D / 128), 256, 0, stream>>>(
      ao_h, WoutT, out, B * T, D, INNER, 1.0f);
}
